// Attention_44160853738116
// MI455X (gfx1250) — matrix operand order repaired and checked by reading, not compiled, not run on hardware
//
#include <hip/hip_runtime.h>
#include <hip/hip_bf16.h>
#include <cstdint>

// ---------------------------------------------------------------------------
// Problem constants (trajectory attention): B=2 clips, F=8 frames, P=196,
// C=768, H=12 heads, D=64. N = P*F = 1568 tokens per clip, Bh = 24.
// ---------------------------------------------------------------------------
constexpr int kBC = 2;            // clips
constexpr int kFR = 8;            // frames
constexpr int kP  = 196;          // tokens per frame
constexpr int kC  = 768;          // model dim
constexpr int kH  = 12;           // heads
constexpr int kD  = 64;           // head dim
constexpr int kBF = kBC * kFR;    // 16  (B*F rows of x)
constexpr int kN  = kP * kFR;     // 1568 tokens per clip
constexpr int kBH = kBC * kH;     // 24 batch-heads
constexpr float kScale = 0.125f;  // D^-0.5

typedef __attribute__((ext_vector_type(16))) __bf16 v16bf;
typedef __attribute__((ext_vector_type(8)))  float  v8f;

// ---------------------------------------------------------------------------
// WMMA fragment helpers (CDNA5 wave32; C/D per CDNA5 ISA §7.12.2, A/B use contiguous k)
// 16-bit A 16x32:  lanes 0-15 : VGPR r -> K=2r,2r+1 (K=0..15)
//                  lanes 16-31: same pattern shifted by +16 (K=16..31).
// 16-bit B 32x16 (code's k order = A's):  lane&15 = column N; lanes 0-15 hold K=0..15 (2 per VGPR),
//                  lanes 16-31 hold K=16..31.
// f32  C/D 16x16:  VGPR r -> row r + 8*(lane>=16), col = lane&15.
// ---------------------------------------------------------------------------
__device__ __forceinline__ int lane_id() { return (int)(threadIdx.x & 31u); }

__device__ __forceinline__ v16bf frag_a_load(const __bf16* base, int ld) {
  const int lane  = lane_id();
  const int m     = lane & 15;
  const int koff  = (lane & 16) ? 16 : 0;
  v16bf f;
#pragma unroll
  for (int r = 0; r < 8; ++r) {
    const int k = koff + 2 * r;
    f[2 * r]     = base[m * ld + k];
    f[2 * r + 1] = base[m * ld + k + 1];
  }
  return f;
}

// B stored as [n][k] (output column over rows, K contiguous)
__device__ __forceinline__ v16bf frag_b_nk(const __bf16* base, int ld) {
  const int lane = lane_id();
  const int n    = lane & 15;
  const int koff = (lane & 16) ? 16 : 0;
  v16bf f;
#pragma unroll
  for (int r = 0; r < 8; ++r) {
    const int k = koff + 2 * r;
    f[2 * r]     = base[n * ld + k];
    f[2 * r + 1] = base[n * ld + k + 1];
  }
  return f;
}

__device__ __forceinline__ v8f wmma_bf16(v16bf a, v16bf b, v8f c) {
  return __builtin_amdgcn_wmma_f32_16x16x32_bf16(false, a, false, b,
                                                 (short)0, c, false, false);
}

// ---------------------------------------------------------------------------
// Generic GEMM: Cout[M,Ncols] = A[M,K] * W[Ncols,K]^T (+ bias), bf16 WMMA,
// fp32 accumulation. M,Ncols multiples of 64; K multiple of 32.
// Block: 128 threads = 4 waves; 64x64 block tile; each wave owns a 32x32
// sub-tile -> 2 A-frags x 2 B-frags = 4 WMMA per wave per K-chunk
// (1:1 fragment-build : WMMA ratio). Next K-chunk is prefetched
// (global_prefetch_b8) while the current chunk computes.
// ---------------------------------------------------------------------------
template <typename TA, typename TC>
__global__ void __launch_bounds__(128) gemm_nt_kernel(
    const TA* __restrict__ A, const float* __restrict__ W,
    const float* __restrict__ bias, TC* __restrict__ Cout,
    int M, int Ncols, int K) {
  __shared__ __bf16 As[64][36];
  __shared__ __bf16 Bs[64][36];

  const int m0   = blockIdx.x * 64;
  const int n0   = blockIdx.y * 64;
  const int tid  = (int)threadIdx.x;
  const int wave = tid >> 5;
  const int wm   = (wave >> 1) * 32;   // wave sub-tile origin (rows)
  const int wn   = (wave & 1) * 32;    // wave sub-tile origin (cols)
  const int lr   = tid >> 1;           // 0..63 : tile row for staging
  const int lc   = (tid & 1) * 16;     // 0 or 16 : K offset for staging

  v8f acc00 = {0.f, 0.f, 0.f, 0.f, 0.f, 0.f, 0.f, 0.f};
  v8f acc01 = acc00, acc10 = acc00, acc11 = acc00;

  for (int k0 = 0; k0 < K; k0 += 32) {
    const TA*    arow = A + (size_t)(m0 + lr) * K + k0 + lc;
    const float* wrow = W + (size_t)(n0 + lr) * K + k0 + lc;
    // prefetch next K-chunk (speculative; dropped if OOB)
    __builtin_prefetch(arow + 32, 0, 1);
    __builtin_prefetch(wrow + 32, 0, 1);
#pragma unroll
    for (int j = 0; j < 16; ++j) {
      As[lr][lc + j] = (__bf16)(float)arow[j];
      Bs[lr][lc + j] = (__bf16)wrow[j];
    }
    __syncthreads();
    const v16bf a0 = frag_a_load(&As[wm][0], 36);
    const v16bf a1 = frag_a_load(&As[wm + 16][0], 36);
    const v16bf b0 = frag_b_nk(&Bs[wn][0], 36);
    const v16bf b1 = frag_b_nk(&Bs[wn + 16][0], 36);
    acc00 = wmma_bf16(a0, b0, acc00);
    acc01 = wmma_bf16(a0, b1, acc01);
    acc10 = wmma_bf16(a1, b0, acc10);
    acc11 = wmma_bf16(a1, b1, acc11);
    __syncthreads();
  }

  const int lane  = tid & 31;
  const int n     = lane & 15;
  const int mhalf = (lane & 16) ? 8 : 0;
#pragma unroll
  for (int i = 0; i < 2; ++i) {
#pragma unroll
    for (int j = 0; j < 2; ++j) {
      const v8f& acc = (i == 0) ? (j == 0 ? acc00 : acc01)
                                : (j == 0 ? acc10 : acc11);
      const int col  = n0 + wn + j * 16 + n;
      const float bv = bias ? bias[col] : 0.f;
#pragma unroll
      for (int rr = 0; rr < 8; ++rr) {
        const int row = m0 + wm + i * 16 + rr + mhalf;
        Cout[(size_t)row * Ncols + col] = (TC)(acc[rr] + bv);
      }
    }
  }
}

// ---------------------------------------------------------------------------
// Scatter QKV GEMM result into (a) the reference qkv output tensor
// (3, b, h, n, d) fp32 and (b) merged (Bh, N, D) bf16 Q/K/V for stage 1.
// ---------------------------------------------------------------------------
__global__ void __launch_bounds__(256) qkv_scatter_kernel(
    const float* __restrict__ qkv_tmp, float* __restrict__ qkv_out,
    __bf16* __restrict__ q_m, __bf16* __restrict__ k_m,
    __bf16* __restrict__ v_m) {
  const int g = (int)(blockIdx.x * 256 + threadIdx.x);
  if (g >= kBF * kP * 3 * kC) return;
  const int i  = g / (3 * kC);        // row in x: bi*196+ni
  const int j  = g % (3 * kC);        // column in [0, 2304)
  const int bi = i / kP, ni = i % kP;
  const int s  = j / kC;              // 0=q, 1=k, 2=v
  const int rm = j % kC;
  const int hi = rm >> 6, di = rm & 63;

  const float v = qkv_tmp[(size_t)g];
  qkv_out[((((size_t)s * kBF + bi) * kH + hi) * kP + ni) * kD + di] = v;

  const int bc = bi >> 3, f = bi & 7;
  const int bh = bc * kH + hi;
  const int t  = f * kP + ni;
  __bf16* dst = (s == 0) ? q_m : (s == 1) ? k_m : v_m;
  dst[((size_t)bh * kN + t) * kD + di] = (__bf16)v;
}

// ---------------------------------------------------------------------------
// Stage-1 spatial attention. One wave per (bh, 16-query tile, frame).
// scores(16x196) = Q(16x64) K^T, fp32 softmax over 196 keys, then
// xo(16x64) = attn(16x224, zero-padded) V(224x64). All matmuls via WMMA.
// Also emits the diagonal-gather rows x_diag[bc][s] = xo[bc][s][s/196].
// ---------------------------------------------------------------------------
__global__ void __launch_bounds__(32) stage1_attn_kernel(
    const __bf16* __restrict__ qm, const __bf16* __restrict__ km,
    const __bf16* __restrict__ vm, __bf16* __restrict__ xo_bf,
    __bf16* __restrict__ xd_bf) {
  __shared__ float  sc[16][208];
  __shared__ __bf16 at[16][224];

  const int qt   = (int)blockIdx.x;  // 0..97
  const int f    = (int)blockIdx.y;  // 0..7
  const int bh   = (int)blockIdx.z;  // 0..23
  const int lane = (int)threadIdx.x;
  const int q0   = qt * 16;
  const int n    = lane & 15;
  const int mhalf  = (lane & 16) ? 8 : 0;
  const int koff16 = (lane & 16) ? 16 : 0;

  // Q fragments (16x64 -> two K=32 chunks), reused across all key tiles
  const __bf16* qbase = qm + ((size_t)bh * kN + q0) * kD;
  const v16bf qa0 = frag_a_load(qbase, kD);
  const v16bf qa1 = frag_a_load(qbase + 32, kD);

  const size_t kvrow0 = (size_t)bh * kN + (size_t)f * kP;

  // ---- scores = scale * Q K^T, masked past kP ----
  for (int kt = 0; kt < 13; ++kt) {
    const int key  = kt * 16 + n;
    const int keyc = (key < kP) ? key : kP - 1;
    const __bf16* kb = km + (kvrow0 + keyc) * kD;
    v16bf b0, b1;
#pragma unroll
    for (int r = 0; r < 8; ++r) {
      const int d = koff16 + 2 * r;
      b0[2 * r]     = kb[d];
      b0[2 * r + 1] = kb[d + 1];
      b1[2 * r]     = kb[32 + d];
      b1[2 * r + 1] = kb[32 + d + 1];
    }
    v8f acc = {0.f, 0.f, 0.f, 0.f, 0.f, 0.f, 0.f, 0.f};
    acc = wmma_bf16(qa0, b0, acc);
    acc = wmma_bf16(qa1, b1, acc);
#pragma unroll
    for (int rr = 0; rr < 8; ++rr)
      sc[rr + mhalf][kt * 16 + n] = (key < kP) ? acc[rr] * kScale : -1e30f;
  }
  __syncthreads();

  // ---- fp32 softmax over 196 keys, emit bf16 attn tile (zero-padded) ----
  if (lane < 16) {
    float mx = -1e30f;
    for (int k = 0; k < kP; ++k) mx = fmaxf(mx, sc[lane][k]);
    float s = 0.f;
    for (int k = 0; k < kP; ++k) {
      const float e = __expf(sc[lane][k] - mx);
      sc[lane][k] = e;
      s += e;
    }
    const float inv = 1.f / s;
    for (int k = 0; k < kP; ++k) at[lane][k] = (__bf16)(sc[lane][k] * inv);
    for (int k = kP; k < 224; ++k) at[lane][k] = (__bf16)0.f;
  }
  __syncthreads();

  // ---- xo = attn @ V ----
  const int bc = bh / kH, h = bh % kH;
  for (int nt = 0; nt < 4; ++nt) {
    v8f acc = {0.f, 0.f, 0.f, 0.f, 0.f, 0.f, 0.f, 0.f};
    for (int kc = 0; kc < 7; ++kc) {
      const v16bf a = frag_a_load(&at[0][kc * 32], 224);
      v16bf b;
#pragma unroll
      for (int r = 0; r < 8; ++r) {
#pragma unroll
        for (int e = 0; e < 2; ++e) {
          const int key  = kc * 32 + koff16 + 2 * r + e;
          const int keyc = (key < kP) ? key : kP - 1;  // attn pad is exactly 0
          b[2 * r + e] = vm[(kvrow0 + keyc) * kD + nt * 16 + n];
        }
      }
      acc = wmma_bf16(a, b, acc);
    }
#pragma unroll
    for (int rr = 0; rr < 8; ++rr) {
      const int q = q0 + rr + mhalf;
      const int c = h * kD + nt * 16 + n;
      const float val = acc[rr];
      xo_bf[(((size_t)bc * kN + q) * kFR + f) * kC + c] = (__bf16)val;
      if (q / kP == f)
        xd_bf[((size_t)bc * kN + q) * kC + c] = (__bf16)val;
    }
  }
}

// ---------------------------------------------------------------------------
// Stage-2 temporal attention (tiny: 8 frames per token) — VALU.
// One block per (bc, s): 96 dot products of length 64, softmax over 8,
// writes attn output tensor and pre-projection rows.
// ---------------------------------------------------------------------------
__global__ void __launch_bounds__(128) stage2_kernel(
    const float* __restrict__ q2, const __bf16* __restrict__ kv,
    float* __restrict__ attn_out, float* __restrict__ out_attn) {
  __shared__ float dots[kH][kFR];
  __shared__ float w[kH][kFR];

  const int idx = (int)blockIdx.x;  // 0..3135
  const int bc = idx / kN, s = idx % kN;
  const int t  = (int)threadIdx.x;

  const float*  qrow  = q2 + ((size_t)bc * kN + s) * kC;
  const __bf16* kvrow = kv + ((size_t)bc * kN + s) * kFR * (2 * kC);

  if (t < kH * kFR) {
    const int h = t >> 3, f = t & 7;
    const float*  qh = qrow + h * kD;
    const __bf16* kh = kvrow + (size_t)f * 2 * kC + h * kD;
    float acc = 0.f;
#pragma unroll
    for (int d = 0; d < kD; ++d) acc += qh[d] * (float)kh[d];
    dots[h][f] = acc * kScale;
  }
  __syncthreads();

  if (t < kH) {
    float mx = -1e30f;
#pragma unroll
    for (int f = 0; f < kFR; ++f) mx = fmaxf(mx, dots[t][f]);
    float e[kFR], sum = 0.f;
#pragma unroll
    for (int f = 0; f < kFR; ++f) {
      e[f] = __expf(dots[t][f] - mx);
      sum += e[f];
    }
    const float inv = 1.f / sum;
#pragma unroll
    for (int f = 0; f < kFR; ++f) w[t][f] = e[f] * inv;
  }
  __syncthreads();

  if (t < kH * kFR) {
    const int h = t >> 3, f = t & 7;
    attn_out[(((size_t)bc * kH + h) * kN + s) * kFR + f] = w[h][f];
  }

  // out row r = (bc*F + s/P)*P + s%P ; col c = h*64 + d
  const int r = (bc * kFR + s / kP) * kP + (s % kP);
  for (int c = t; c < kC; c += 128) {
    const int h = c >> 6;
    float o = 0.f;
#pragma unroll
    for (int f = 0; f < kFR; ++f)
      o += w[h][f] * (float)kvrow[(size_t)f * 2 * kC + kC + c];
    out_attn[(size_t)r * kC + c] = o;
  }
}

// ---------------------------------------------------------------------------
// Launch
// ---------------------------------------------------------------------------
extern "C" void kernel_launch(void* const* d_in, const int* in_sizes, int n_in,
                              void* d_out, int out_size, void* d_ws,
                              size_t ws_size, hipStream_t stream) {
  (void)in_sizes; (void)n_in; (void)out_size; (void)ws_size;

  const float* x      = (const float*)d_in[0];
  const float* W_qkv  = (const float*)d_in[1];
  const float* W_q    = (const float*)d_in[2];
  const float* W_kv   = (const float*)d_in[3];
  const float* W_proj = (const float*)d_in[4];
  const float* b_proj = (const float*)d_in[5];

  // Output tuple: out (16,196,768) | attn (2,12,1568,8) | qkv (3,16,12,196,64)
  float* out0     = (float*)d_out;
  float* attn_out = out0 + (size_t)kBF * kP * kC;
  float* qkv_out  = attn_out + (size_t)kBC * kH * kN * kFR;

  // Workspace carve-out (~183 MB)
  char* ws = (char*)d_ws;
  size_t off = 0;
  auto alloc = [&](size_t bytes) -> void* {
    void* p = ws + off;
    off += (bytes + 255) & ~(size_t)255;
    return p;
  };
  const int Mx = kBF * kP;  // 3136 rows of x / out
  float*  qkv_tmp = (float*) alloc((size_t)Mx * 3 * kC * 4);
  __bf16* q_m     = (__bf16*)alloc((size_t)kBH * kN * kD * 2);
  __bf16* k_m     = (__bf16*)alloc((size_t)kBH * kN * kD * 2);
  __bf16* v_m     = (__bf16*)alloc((size_t)kBH * kN * kD * 2);
  __bf16* xo_bf   = (__bf16*)alloc((size_t)kBC * kN * kFR * kC * 2);
  __bf16* xd_bf   = (__bf16*)alloc((size_t)kBC * kN * kC * 2);
  float*  q2      = (float*) alloc((size_t)kBC * kN * kC * 4);
  __bf16* kv      = (__bf16*)alloc((size_t)kBC * kN * kFR * 2 * kC * 2);
  float*  out_pre = (float*) alloc((size_t)Mx * kC * 4);

  // 1) QKV projection: (3136,768) @ (2304,768)^T
  gemm_nt_kernel<float, float>
      <<<dim3(Mx / 64, (3 * kC) / 64), 128, 0, stream>>>(
          x, W_qkv, nullptr, qkv_tmp, Mx, 3 * kC, kC);

  // 1b) scatter to qkv output tensor + merged bf16 q/k/v
  const int tot = Mx * 3 * kC;
  qkv_scatter_kernel<<<(tot + 255) / 256, 256, 0, stream>>>(
      qkv_tmp, qkv_out, q_m, k_m, v_m);

  // 2) Stage-1 spatial attention (per batch-head, per frame)
  stage1_attn_kernel<<<dim3(kN / 16, kFR, kBH), 32, 0, stream>>>(
      q_m, k_m, v_m, xo_bf, xd_bf);

  // 3) kv = xo @ W_kv^T : (25088,768) @ (1536,768)^T -> bf16
  gemm_nt_kernel<__bf16, __bf16>
      <<<dim3((kBC * kN * kFR) / 64, (2 * kC) / 64), 128, 0, stream>>>(
          xo_bf, W_kv, nullptr, kv, kBC * kN * kFR, 2 * kC, kC);

  // 4) q2 = x_diag @ W_q^T : (3136,768) @ (768,768)^T -> f32
  gemm_nt_kernel<__bf16, float>
      <<<dim3((kBC * kN) / 64, kC / 64), 128, 0, stream>>>(
          xd_bf, W_q, nullptr, q2, kBC * kN, kC, kC);

  // 5) Stage-2 temporal attention + attn output
  stage2_kernel<<<kBC * kN, 128, 0, stream>>>(q2, kv, attn_out, out_pre);

  // 6) Final projection: out = out_pre @ W_proj^T + b_proj
  gemm_nt_kernel<float, float>
      <<<dim3(Mx / 64, kC / 64), 128, 0, stream>>>(
          out_pre, W_proj, b_proj, out0, Mx, kC, kC);
}
